// DeformationReasoning_28750511079685
// MI455X (gfx1250) — matrix operand order repaired and checked by reading, not compiled, not run on hardware
//
#include <hip/hip_runtime.h>
#include <hip/hip_bf16.h>

// ---------------------------------------------------------------------------
// DeformationReasoning for MI455X (gfx1250): bf16 WMMA 16x16x32 pipeline.
//   per-block (per mesh vertex n):
//     X(48x352 bf16, padded from 43x339) in LDS
//     step1: P_k[o][j] = (X @ W_k)^T       (v_wmma_f32_16x16x32_bf16)
//     step2: X = relu(sum_k adj_k @ P_k + b) (+residual)
//     ... 6 layers, then softmax-weighted coord update.
// ---------------------------------------------------------------------------

typedef __attribute__((ext_vector_type(16))) __bf16 v16bf;
typedef __attribute__((ext_vector_type(8)))  __bf16 v8bf;
typedef __attribute__((ext_vector_type(8)))  float  v8f;

#define S_   43
#define SP   48     // S padded to 3 * 16
#define F_   339
#define FP   352    // F padded to 11 * 32
#define H_   192    // 12 * 16
#define KSUP 2
#define JP   64     // adjacency K-dim padded to 2 * 32

// ---- WMMA fragment loaders (CDNA5 16-bit A/B VGPR layouts, wave32) --------
// A (MxK=16x32): lane L: row = L%16; halves 0..7  -> K = kb + (L/16)*8 + 0..7
//                                   halves 8..15 -> K = kb + 16 + (L/16)*8 + 0..7
static __device__ __forceinline__ v16bf load_frag_a(const __bf16* base, int pitch,
                                                    int row, int kb, int lh) {
  const __bf16* p = base + row * pitch + kb + lh * 8;
  v8bf lo = *(const v8bf*)p;
  v8bf hi = *(const v8bf*)(p + 16);
  v16bf r;
#pragma unroll
  for (int i = 0; i < 8; ++i) { r[i] = lo[i]; r[i + 8] = hi[i]; }
  return r;
}

// B (KxN=32x16) stored column-major (i.e. row-major W^T [col][K]):
// lane L: col = L%16; halves 0..7 -> K = kb + (L/16)*8 + 0..7; 8..15 -> K = kb + 16 + (L/16)*8 + 0..7
static __device__ __forceinline__ v16bf load_frag_b(const __bf16* base, int pitch,
                                                    int col, int kb, int lh) {
  const __bf16* p = base + col * pitch + kb + lh * 8;
  v8bf lo = *(const v8bf*)p;
  v8bf hi = *(const v8bf*)(p + 16);
  v16bf r;
#pragma unroll
  for (int i = 0; i < 8; ++i) { r[i] = lo[i]; r[i + 8] = hi[i]; }
  return r;
}

static __device__ __forceinline__ v8f wmma_bf16(v16bf a, v16bf b, v8f c) {
  return __builtin_amdgcn_wmma_f32_16x16x32_bf16(false, a, false, b,
                                                 (short)0, c, false, false);
}

// step1: P_k = (X(48 x Kd) @ W_k)^T, stored P[o][j] (pitch JP) as bf16.
static __device__ __forceinline__ void step1(const __bf16* X, int Kd,
                                             const __bf16* Wt0, const __bf16* Wt1,
                                             __bf16* P0, __bf16* P1,
                                             int wave, int l16, int lh) {
  for (int job = wave; job < 2 * 3 * 12; job += 8) {
    const int k   = job >= 36;
    const int rem = job - k * 36;
    const int tm  = rem / 12;
    const int tn  = rem % 12;
    const __bf16* Wt = k ? Wt1 : Wt0;
    __bf16*       P  = k ? P1  : P0;
    const int row = tm * 16 + l16;
    const int col = tn * 16 + l16;
    v8f acc = {0.f, 0.f, 0.f, 0.f, 0.f, 0.f, 0.f, 0.f};
    for (int kb = 0; kb < Kd; kb += 32) {
      v16bf a = load_frag_a(X, Kd, row, kb, lh);
      v16bf b = load_frag_b(Wt, Kd, col, kb, lh);
      acc = wmma_bf16(a, b, acc);
    }
    // transposed store: lane's 8 accumulator rows are contiguous in j.
    v8bf d;
#pragma unroll
    for (int r = 0; r < 8; ++r) d[r] = (__bf16)acc[r];
    *(v8bf*)(P + col * JP + tm * 16 + lh * 8) = d;
  }
}

// step2: X(48 x 192) = relu(sum_k adj_k(48x64) @ P_k + bias) [+R] [->R]
static __device__ __forceinline__ void step2(const __bf16* Aj,
                                             const __bf16* P0, const __bf16* P1,
                                             const float* bias,
                                             __bf16* X, __bf16* R,
                                             bool addRes, bool saveRes,
                                             int wave, int l16, int lh) {
  for (int job = wave; job < 3 * 12; job += 8) {
    const int tm = job / 12;
    const int tn = job % 12;
    const int row = tm * 16 + l16;
    const int col = tn * 16 + l16;
    v8f acc = {0.f, 0.f, 0.f, 0.f, 0.f, 0.f, 0.f, 0.f};
#pragma unroll
    for (int k = 0; k < KSUP; ++k) {
      const __bf16* A = Aj + k * SP * JP;
      const __bf16* P = k ? P1 : P0;
#pragma unroll
      for (int kb = 0; kb < JP; kb += 32) {
        v16bf a = load_frag_a(A, JP, row, kb, lh);
        v16bf b = load_frag_b(P, JP, col, kb, lh);
        acc = wmma_bf16(a, b, acc);
      }
    }
    const float bv = bias[col];
    const int i0 = tm * 16 + lh * 8;
#pragma unroll
    for (int r = 0; r < 8; ++r) {
      float v = fmaxf(acc[r] + bv, 0.f);
      if (addRes) v += (float)R[(i0 + r) * H_ + col];
      __bf16 h = (__bf16)v;
      X[(i0 + r) * H_ + col] = h;
      if (saveRes) R[(i0 + r) * H_ + col] = h;
    }
  }
}

// ---------------- weight / adjacency prep (fp32 -> bf16, transpose+pad) ----
__global__ void prep_w1(const float* __restrict__ w1, __bf16* __restrict__ w1t) {
  int idx = blockIdx.x * 256 + threadIdx.x;
  if (idx >= KSUP * H_ * FP) return;
  int f = idx % FP;
  int o = (idx / FP) % H_;
  int k = idx / (FP * H_);
  float v = (f < F_) ? w1[(k * F_ + f) * H_ + o] : 0.f;
  w1t[idx] = (__bf16)v;
}

__global__ void prep_wmid(const float* __restrict__ wm, __bf16* __restrict__ wt) {
  int idx = blockIdx.x * 256 + threadIdx.x;
  if (idx >= 4 * KSUP * H_ * H_) return;
  int j  = idx % H_;
  int o  = (idx / H_) % H_;
  int kk = idx / (H_ * H_);          // kk = layer*2 + k
  wt[idx] = (__bf16)wm[(kk * H_ + j) * H_ + o];
}

__global__ void prep_adj(const float* __restrict__ adj, __bf16* __restrict__ a) {
  int idx = blockIdx.x * 256 + threadIdx.x;
  if (idx >= KSUP * SP * JP) return;
  int j = idx % JP;
  int i = (idx / JP) % SP;
  int k = idx / (JP * SP);
  float v = (i < S_ && j < S_) ? adj[(k * S_ + i) * S_ + j] : 0.f;
  a[idx] = (__bf16)v;
}

// ---------------------------- main kernel ----------------------------------
__global__ __launch_bounds__(256, 1)
void deform_kernel(const float* __restrict__ proj_feat,
                   const float* __restrict__ prev_coord,
                   const float* __restrict__ adjf,
                   const float* __restrict__ coord,
                   const float* __restrict__ b1,
                   const float* __restrict__ bmid,
                   const float* __restrict__ w6,
                   const float* __restrict__ b6,
                   const __bf16* __restrict__ w1t,
                   const __bf16* __restrict__ wmidt,
                   const __bf16* __restrict__ adjp,
                   float* __restrict__ out) {
  const int tid  = threadIdx.x;
  const int wave = tid >> 5;
  const int lane = tid & 31;
  const int l16  = lane & 15;
  const int lh   = lane >> 4;
  const int n    = blockIdx.x;

  __shared__ __align__(16) __bf16 Xs[SP * FP];        // 33.0 KiB (layer1 input)
  __shared__ __align__(16) __bf16 P0[H_ * JP];        // 24 KiB
  __shared__ __align__(16) __bf16 P1[H_ * JP];        // 24 KiB
  __shared__ __align__(16) __bf16 Rs[SP * H_];        // 18 KiB (residual)
  __shared__ __align__(16) __bf16 Aj[KSUP * SP * JP]; // 12 KiB
  __shared__ float tb[KSUP * S_];
  __shared__ float x6s[S_];

  // adjacency (already bf16, padded) into LDS
  for (int i = tid; i < KSUP * SP * JP; i += 256) Aj[i] = adjp[i];
  // zero the never-written pad rows j=48..63 of P0/P1 once
  for (int i = tid; i < H_ * 16; i += 256) {
    int o = i / 16, j = SP + (i % 16);
    P0[o * JP + j] = (__bf16)0.f;
    P1[o * JP + j] = (__bf16)0.f;
  }
  // layer-1 input: fp32 -> bf16 with padding (43x339 -> 48x352)
  const float* xin = proj_feat + (size_t)n * S_ * F_;
  for (int i = tid; i < SP * FP; i += 256) {
    int r = i / FP, c = i % FP;
    float v = (r < S_ && c < F_) ? xin[r * F_ + c] : 0.f;
    Xs[i] = (__bf16)v;
  }
  __syncthreads();

  // ---- layer 1 : F=352 -> H, save x1 as residual
  step1(Xs, FP, w1t, w1t + (size_t)H_ * FP, P0, P1, wave, l16, lh);
  __syncthreads();
  step2(Aj, P0, P1, b1, Xs, Rs, false, true, wave, l16, lh);
  __syncthreads();

  // ---- 4 mid layers : H -> H
  // l=0: plain; l=1: +x1, save x3; l=2: plain; l=3: +x3
  const bool addR[4]  = {false, true,  false, true};
  const bool saveR[4] = {false, true,  false, false};
#pragma unroll 1
  for (int l = 0; l < 4; ++l) {
    const __bf16* Wt0 = wmidt + (size_t)(l * 2 + 0) * H_ * H_;
    const __bf16* Wt1 = wmidt + (size_t)(l * 2 + 1) * H_ * H_;
    step1(Xs, H_, Wt0, Wt1, P0, P1, wave, l16, lh);
    __syncthreads();
    step2(Aj, P0, P1, bmid + l * H_, Xs, Rs, addR[l], saveR[l], wave, l16, lh);
    __syncthreads();
  }

  // ---- layer 6 (H -> 1): t[k][j] = <x5[j,:], w6[k,:,0]>
  if (tid < KSUP * S_) {
    int k = tid / S_, j = tid % S_;
    const float* wv = w6 + k * H_;
    float s = 0.f;
    for (int f = 0; f < H_; ++f) s += (float)Xs[j * H_ + f] * wv[f];
    tb[tid] = s;
  }
  __syncthreads();
  // x6[i] = relu( sum_k sum_j adj[k,i,j] * t[k][j] + b6 )
  if (tid < S_) {
    float s = 0.f;
    for (int k = 0; k < KSUP; ++k) {
      const float* ar = adjf + (k * S_ + tid) * S_;
      const float* tv = tb + k * S_;
      for (int j = 0; j < S_; ++j) s += ar[j] * tv[j];
    }
    x6s[tid] = fmaxf(s + b6[0], 0.f);
  }
  __syncthreads();
  // softmax over S + weighted coordinate blend (tiny, serial on thread 0)
  if (tid == 0) {
    float m = -1e30f;
    for (int i = 0; i < S_; ++i) m = fmaxf(m, x6s[i]);
    const float* cp = coord + (size_t)n * S_ * 3;
    float den = 0.f, a0 = 0.f, a1 = 0.f, a2 = 0.f;
    for (int i = 0; i < S_; ++i) {
      float e = __expf(x6s[i] - m);
      den += e;
      a0 += e * cp[i * 3 + 0];
      a1 += e * cp[i * 3 + 1];
      a2 += e * cp[i * 3 + 2];
    }
    float inv = 1.f / den;
    out[n * 3 + 0] = a0 * inv + prev_coord[n * 3 + 0];
    out[n * 3 + 1] = a1 * inv + prev_coord[n * 3 + 1];
    out[n * 3 + 2] = a2 * inv + prev_coord[n * 3 + 2];
  }
}

// ---------------------------- launcher -------------------------------------
extern "C" void kernel_launch(void* const* d_in, const int* in_sizes, int n_in,
                              void* d_out, int out_size, void* d_ws, size_t ws_size,
                              hipStream_t stream) {
  const float* proj_feat = (const float*)d_in[0];
  const float* prev      = (const float*)d_in[1];
  const float* adjf      = (const float*)d_in[2];
  const float* coord     = (const float*)d_in[3];
  const float* w1        = (const float*)d_in[4];
  const float* b1        = (const float*)d_in[5];
  const float* wmid      = (const float*)d_in[6];
  const float* bmid      = (const float*)d_in[7];
  const float* w6        = (const float*)d_in[8];
  const float* b6        = (const float*)d_in[9];

  const size_t w1t_bytes  = (size_t)KSUP * H_ * FP * 2;       // 270336
  const size_t wmid_bytes = (size_t)4 * KSUP * H_ * H_ * 2;   // 589824
  __bf16* w1t   = (__bf16*)d_ws;
  __bf16* wmidt = (__bf16*)((char*)d_ws + w1t_bytes);
  __bf16* adjp  = (__bf16*)((char*)d_ws + w1t_bytes + wmid_bytes);

  const int N = in_sizes[0] / (S_ * F_);

  prep_w1  <<<(KSUP * H_ * FP      + 255) / 256, 256, 0, stream>>>(w1,   w1t);
  prep_wmid<<<(4 * KSUP * H_ * H_  + 255) / 256, 256, 0, stream>>>(wmid, wmidt);
  prep_adj <<<(KSUP * SP * JP      + 255) / 256, 256, 0, stream>>>(adjf, adjp);

  deform_kernel<<<N, 256, 0, stream>>>(proj_feat, prev, adjf, coord,
                                       b1, bmid, w6, b6,
                                       w1t, wmidt, adjp, (float*)d_out);
}
